// _GNNWrapper_28656021799028
// MI455X (gfx1250) — matrix operand order repaired and checked by reading, not compiled, not run on hardware
//
#include <hip/hip_runtime.h>
#include <hip/hip_fp16.h>
#include <math.h>

typedef __attribute__((ext_vector_type(16))) _Float16 v16h;
typedef __attribute__((ext_vector_type(8)))  _Float16 v8h;
typedef __attribute__((ext_vector_type(8)))  float    v8f;

#define N_NODES 100000
#define N_EDGES 1600000
#define HDIM    128
#define VDIM    64

// ---------------- degree: atomic count of in-edges ----------------
__global__ void deg_kernel(const long long* __restrict__ ei, float* __restrict__ deg) {
    int e = blockIdx.x * blockDim.x + threadIdx.x;
    if (e < N_EDGES) {
        int dst = (int)ei[N_EDGES + e];
        atomicAdd(&deg[dst], 1.0f);
    }
}

// ---------------- layer-0 scalar aggregation ----------------
__global__ void agg0_kernel(const long long* __restrict__ ei,
                            const float* __restrict__ x,
                            float* __restrict__ agg0) {
    int e = blockIdx.x * blockDim.x + threadIdx.x;
    if (e < N_EDGES) {
        int src = (int)ei[e];
        int dst = (int)ei[N_EDGES + e];
        atomicAdd(&agg0[dst], x[src]);
    }
}

// ---------------- layer 0: (N,1) -> (N,128), relu ----------------
__global__ void layer0_kernel(const float* __restrict__ x,
                              const float* __restrict__ agg0,
                              const float* __restrict__ deg,
                              const float* __restrict__ Wl0,
                              const float* __restrict__ bl0,
                              const float* __restrict__ Wr0,
                              float* __restrict__ h) {
    int idx = blockIdx.x * blockDim.x + threadIdx.x;
    if (idx < N_NODES * HDIM) {
        int n = idx >> 7;
        int j = idx & (HDIM - 1);
        float invd = 1.0f / fmaxf(deg[n], 1.0f);
        float v = agg0[n] * invd * Wl0[j] + bl0[j] + x[n] * Wr0[j];
        h[idx] = fmaxf(v, 0.0f);
    }
}

// ------- pre-convert weights: wt[layer][col(128)][k(256)] f16, k<128 -> Wl, else Wr -------
__global__ void wprep_kernel(const float* __restrict__ Wl,
                             const float* __restrict__ Wr,
                             _Float16* __restrict__ wt) {
    int idx = blockIdx.x * blockDim.x + threadIdx.x;   // 2*128*256 = 65536
    if (idx < 2 * 128 * 256) {
        int layer = idx >> 15;
        int j = (idx >> 8) & 127;
        int k = idx & 255;
        float v = (k < HDIM) ? Wl[layer * HDIM * HDIM + k * HDIM + j]
                             : Wr[layer * HDIM * HDIM + (k - HDIM) * HDIM + j];
        wt[idx] = (_Float16)v;
    }
}

// ---------------- edge scatter: agg[dst] += h[src], 128 floats/edge ----------------
__global__ void scatter_kernel(const long long* __restrict__ ei,
                               const float* __restrict__ h,
                               float* __restrict__ agg) {
    long long idx = (long long)blockIdx.x * blockDim.x + threadIdx.x;
    long long e = idx >> 5;
    if (e < N_EDGES) {
        int lane = (int)(idx & 31);
        int src = (int)ei[e];
        int dst = (int)ei[N_EDGES + e];
        const float4 v = *(const float4*)(h + (size_t)src * HDIM + lane * 4);
        float* a = agg + (size_t)dst * HDIM + lane * 4;
        atomicAdd(a + 0, v.x);
        atomicAdd(a + 1, v.y);
        atomicAdd(a + 2, v.z);
        atomicAdd(a + 3, v.w);
    }
}

// ---- fused SAGE GEMM: out = relu([agg/deg | h] (16x256) @ wt^T (256x128) + bias) ----
// 256 threads = 8 waves; wave w computes a 16x16 tile at columns [16w,16w+16).
// A staged in LDS as f16, row stride 264 halves (bank-conflict padding).
__global__ __launch_bounds__(256) void sage_gemm_kernel(
    const float* __restrict__ agg, const float* __restrict__ deg,
    const float* __restrict__ h_in, const _Float16* __restrict__ wt,
    const float* __restrict__ bias, float* __restrict__ h_out)
{
    __shared__ _Float16 Atile[16 * 264];
    const int tid  = threadIdx.x;
    const int tile = blockIdx.x;

    // ---- stage A = [agg/deg (k<128) | h (k>=128)] as f16 ----
    {
        int row = tid >> 4;
        int k0  = (tid & 15) * 16;
        int g   = tile * 16 + row;
        float invd = 1.0f / fmaxf(deg[g], 1.0f);
        const float* ar = agg  + (size_t)g * HDIM;
        const float* hr = h_in + (size_t)g * HDIM;
#pragma unroll
        for (int i = 0; i < 16; ++i) {
            int k = k0 + i;
            float v = (k < HDIM) ? ar[k] * invd : hr[k - HDIM];
            Atile[row * 264 + k] = (_Float16)v;
        }
    }
    __syncthreads();

    const int wv    = tid >> 5;
    const int lane  = tid & 31;
    const int lhalf = lane >> 4;        // 0: lanes 0-15, 1: lanes 16-31
    const int l15   = lane & 15;
    const int col   = wv * 16 + l15;
    const _Float16* wcol = wt + (size_t)col * 256;

    v8f c = {};
#pragma unroll
    for (int kb = 0; kb < 256; kb += 32) {
        // A fragment (16x32 f16): lane row = l15; K halves [kb+8*lhalf .. +7] and [+16 .. +23]
        const _Float16* arow = &Atile[l15 * 264 + kb + lhalf * 8];
        v8h a0 = *(const v8h*)(arow);
        v8h a1 = *(const v8h*)(arow + 16);
        // B fragment (32x16 f16): lane col = col; K halves [kb+8*lhalf .. +7] and [+16 .. +23]
        const v8h* bp = (const v8h*)(wcol + kb + lhalf * 8);
        v8h b0 = bp[0];
        v8h b1 = bp[2];
        v16h a, b;
#pragma unroll
        for (int i = 0; i < 8; ++i) {
            a[i] = a0[i]; a[8 + i] = a1[i];
            b[i] = b0[i]; b[8 + i] = b1[i];
        }
        c = __builtin_amdgcn_wmma_f32_16x16x32_f16(
                /*neg_a=*/false, a, /*neg_b=*/false, b,
                /*c_mod=*/(short)0, c, /*reuse_a=*/false, /*reuse_b=*/false);
    }

    float bcol = bias[col];
#pragma unroll
    for (int v = 0; v < 8; ++v) {
        int m = v + lhalf * 8;               // D layout: VGPR v -> row v (+8 for lanes 16-31)
        int g = tile * 16 + m;
        h_out[(size_t)g * HDIM + col] = fmaxf(c[v] + bcol, 0.0f);
    }
}

// ---------------- global mean pool (sum, divided in head) ----------------
#define POOL_ROWS 512
__global__ void pool_kernel(const float* __restrict__ h, float* __restrict__ pooled) {
    int j  = threadIdx.x;                 // 128 threads = columns
    int r0 = blockIdx.x * POOL_ROWS;
    int r1 = r0 + POOL_ROWS; if (r1 > N_NODES) r1 = N_NODES;
    float s = 0.0f;
    for (int r = r0; r < r1; ++r) s += h[(size_t)r * HDIM + j];
    atomicAdd(&pooled[j], s);
}

// ---------------- MLP head: z=pooled@Wv+bv, concat [T,t], relu(.@Wm1+bm1)@Wm2+bm2, softplus ----
__global__ void head_kernel(const float* __restrict__ pooled,
                            const float* __restrict__ Wv, const float* __restrict__ bv,
                            const float* __restrict__ temp, const float* __restrict__ tt,
                            const float* __restrict__ Wm1, const float* __restrict__ bm1,
                            const float* __restrict__ Wm2, const float* __restrict__ bm2,
                            float* __restrict__ out) {
    __shared__ float feat[VDIM + 2];
    __shared__ float m1[HDIM];
    __shared__ float red[HDIM];
    int tid = threadIdx.x;                // 128 threads
    const float invN = 1.0f / (float)N_NODES;
    if (tid < VDIM) {
        float s = bv[tid];
        for (int k = 0; k < HDIM; ++k) s += (pooled[k] * invN) * Wv[k * VDIM + tid];
        feat[tid] = s;
    }
    if (tid == 0) { feat[VDIM] = temp[0]; feat[VDIM + 1] = tt[0]; }
    __syncthreads();
    {
        float s = bm1[tid];
        for (int k = 0; k < VDIM + 2; ++k) s += feat[k] * Wm1[k * HDIM + tid];
        m1[tid] = fmaxf(s, 0.0f);
    }
    __syncthreads();
    red[tid] = m1[tid] * Wm2[tid];
    __syncthreads();
    for (int off = 64; off > 0; off >>= 1) {
        if (tid < off) red[tid] += red[tid + off];
        __syncthreads();
    }
    if (tid == 0) {
        float xx = red[0] + bm2[0];
        float sp = (xx > 20.0f) ? xx : ((xx < -20.0f) ? expf(xx) : log1pf(expf(xx)));
        out[0] = sp;
    }
}

extern "C" void kernel_launch(void* const* d_in, const int* in_sizes, int n_in,
                              void* d_out, int out_size, void* d_ws, size_t ws_size,
                              hipStream_t stream) {
    const float*     x    = (const float*)d_in[0];
    const long long* ei   = (const long long*)d_in[1];
    // d_in[2] = batch (all zeros) unused
    const float* temp = (const float*)d_in[3];
    const float* tt   = (const float*)d_in[4];
    const float* Wl0  = (const float*)d_in[5];
    const float* bl0  = (const float*)d_in[6];
    const float* Wr0  = (const float*)d_in[7];
    const float* Wl   = (const float*)d_in[8];
    const float* bl   = (const float*)d_in[9];
    const float* Wr   = (const float*)d_in[10];
    const float* Wv   = (const float*)d_in[11];
    const float* bv   = (const float*)d_in[12];
    const float* Wm1  = (const float*)d_in[13];
    const float* bm1  = (const float*)d_in[14];
    const float* Wm2  = (const float*)d_in[15];
    const float* bm2  = (const float*)d_in[16];
    float* out = (float*)d_out;

    char* ws = (char*)d_ws;
    size_t off = 0;
    auto alloc = [&](size_t bytes) -> void* {
        void* p = ws + off;
        off += (bytes + 255) & ~(size_t)255;
        return p;
    };
    float*     deg    = (float*)alloc((size_t)N_NODES * 4);
    float*     agg0   = (float*)alloc((size_t)N_NODES * 4);
    float*     h      = (float*)alloc((size_t)N_NODES * HDIM * 4);
    float*     agg    = (float*)alloc((size_t)N_NODES * HDIM * 4);
    _Float16*  wt     = (_Float16*)alloc((size_t)2 * 128 * 256 * 2);
    float*     pooled = (float*)alloc(128 * 4);

    hipMemsetAsync(deg,    0, (size_t)N_NODES * 4, stream);
    hipMemsetAsync(agg0,   0, (size_t)N_NODES * 4, stream);
    hipMemsetAsync(pooled, 0, 128 * 4, stream);

    const int TB = 256;
    deg_kernel <<<(N_EDGES + TB - 1) / TB, TB, 0, stream>>>(ei, deg);
    agg0_kernel<<<(N_EDGES + TB - 1) / TB, TB, 0, stream>>>(ei, x, agg0);
    layer0_kernel<<<(N_NODES * HDIM + TB - 1) / TB, TB, 0, stream>>>(
        x, agg0, deg, Wl0, bl0, Wr0, h);
    wprep_kernel<<<(2 * 128 * 256) / TB, TB, 0, stream>>>(Wl, Wr, wt);

    for (int layer = 0; layer < 2; ++layer) {
        hipMemsetAsync(agg, 0, (size_t)N_NODES * HDIM * 4, stream);
        long long sthreads = (long long)N_EDGES * 32;
        scatter_kernel<<<(unsigned)((sthreads + TB - 1) / TB), TB, 0, stream>>>(ei, h, agg);
        sage_gemm_kernel<<<N_NODES / 16, 256, 0, stream>>>(
            agg, deg, h, wt + (size_t)layer * 128 * 256, bl + (size_t)layer * HDIM, h);
    }

    pool_kernel<<<(N_NODES + POOL_ROWS - 1) / POOL_ROWS, 128, 0, stream>>>(h, pooled);
    head_kernel<<<1, 128, 0, stream>>>(pooled, Wv, bv, temp, tt, Wm1, bm1, Wm2, bm2, out);
}
